// FunnelAttention_65034394796619
// MI455X (gfx1250) — matrix operand order repaired and checked by reading, not compiled, not run on hardware
//
#include <hip/hip_runtime.h>

// ---------------------------------------------------------------------------
// FunnelAttention fused kernel for gfx1250 (MI455X)
//   pooled = avgpool(x, 4);  attn = pooled @ x^T;  out = pooled + attn @ x
// bf16 WMMA (v_wmma_f32_16x16x32_bf16) + f32 accumulation, flash-style fusion
// (attn never hits HBM). A transposed bf16 copy of x makes every WMMA operand
// a contiguous b128 load. M=32 per workgroup so each B operand feeds 2 WMMAs.
// ---------------------------------------------------------------------------

typedef __attribute__((ext_vector_type(16))) __bf16        v16bf;
typedef __attribute__((ext_vector_type(8)))  float         v8f;
typedef __attribute__((ext_vector_type(4)))  unsigned int  v4u;

union ABOp { v4u u2[2]; v16bf bf; };

constexpr int Bb    = 4;
constexpr int T     = 4096;
constexpr int C     = 1024;
constexpr int POOLK = 4;
constexpr int P     = T / POOLK;   // 1024

__device__ __forceinline__ unsigned short f2bf(float f) {
  union { __bf16 b; unsigned short s; } u;
  u.b = (__bf16)f;
  return u.s;
}

// ---------------------------------------------------------------------------
// Kernel 1: pooled (f32) + pooled (bf16 shadow)
// ---------------------------------------------------------------------------
__global__ void __launch_bounds__(256) pool_kernel(
    const float* __restrict__ x,
    float* __restrict__ pooled_f,
    unsigned short* __restrict__ pooled_bf) {
  size_t idx = (size_t)blockIdx.x * 256 + threadIdx.x;   // over Bb*P*C
  int    c   = (int)(idx % C);
  size_t bp  = idx / C;
  int    p   = (int)(bp % P);
  int    b   = (int)(bp / P);
  const float* xp = x + ((size_t)b * T + (size_t)p * POOLK) * C + c;
  float s = 0.f;
#pragma unroll
  for (int j = 0; j < POOLK; ++j) s += xp[(size_t)j * C];
  s *= 0.25f;
  pooled_f[idx]  = s;
  pooled_bf[idx] = f2bf(s);
}

// ---------------------------------------------------------------------------
// Kernel 2: x -> bf16 row-major copy AND bf16 transposed copy (LDS-tiled).
// 64x64 tile per workgroup, 256 threads.
// ---------------------------------------------------------------------------
__global__ void __launch_bounds__(256) cvt_transpose(
    const float* __restrict__ x,
    unsigned short* __restrict__ x_bf,
    unsigned short* __restrict__ xT_bf) {
  __shared__ unsigned short tile[64][72];   // pad 72 to break bank alignment
  const int b  = blockIdx.z;
  const int t0 = blockIdx.x * 64;
  const int c0 = blockIdx.y * 64;
  const int tid = threadIdx.x;
  {
    const int tr = tid >> 2;
    const int cb = (tid & 3) * 16;
    const float* src = x + ((size_t)(b * T + t0 + tr)) * C + c0 + cb;
    union { unsigned short s[16]; v4u u[2]; } tmp;
#pragma unroll
    for (int i = 0; i < 16; ++i) {
      unsigned short v = f2bf(src[i]);
      tmp.s[i] = v;
      tile[tr][cb + i] = v;
    }
    v4u* dst = (v4u*)(x_bf + ((size_t)(b * T + t0 + tr)) * C + c0 + cb);
    dst[0] = tmp.u[0];
    dst[1] = tmp.u[1];
  }
  __syncthreads();
  {
    const int cr  = tid >> 2;
    const int tb2 = (tid & 3) * 16;
    union { unsigned short s[16]; v4u u[2]; } tmp;
#pragma unroll
    for (int i = 0; i < 16; ++i) tmp.s[i] = tile[tb2 + i][cr];
    v4u* dst = (v4u*)(xT_bf + ((size_t)(b * C + c0 + cr)) * T + t0 + tb2);
    dst[0] = tmp.u[0];
    dst[1] = tmp.u[1];
  }
}

// ---------------------------------------------------------------------------
// Main fused kernel. Grid: (P/32, Bb). Block: 256 threads = 8 waves (wave32).
// Workgroup tile: M = 32 pooled rows x N = C = 1024 columns.
// Per 256-wide T-block:
//   phase1: each wave computes 4 S-tiles (2 p-tiles x 2 t-tiles) over full
//           K=C; B (x rows) loaded once per k-step feeds both p-tiles.
//           S (32x256) is written bf16 to LDS (row stride 264 halfwords ->
//           conflict-free ds_load_b128 A-operand reads).
//   phase2: each wave owns 8 c-tiles x 2 p-tiles; each B (xT) load feeds
//           2 WMMAs; A (S) comes from LDS once per k-step.
// ---------------------------------------------------------------------------
__global__ void __launch_bounds__(256) funnel_main(
    const float* __restrict__ pooled_f,
    const unsigned short* __restrict__ pooled_bf,
    const unsigned short* __restrict__ x_bf,
    const unsigned short* __restrict__ xT_bf,
    float* __restrict__ out) {
  constexpr int SROW = 264;                       // 32x256 S block, padded rows
  __shared__ unsigned short s_lds[32 * SROW];

  const int tid  = threadIdx.x;
  const int lane = tid & 31;
  const int wv   = tid >> 5;       // wave id 0..7
  const int hl   = lane >> 4;      // lane half (0/1)
  const int lm   = lane & 15;
  const int kbA  = hl * 8;         // A-operand K base (16-bit A layout)
  const int kbB  = hl * 16;        // unused; B-operand loads also use kbA

  const int b     = blockIdx.y;
  const int pbase = blockIdx.x * 32;

  const unsigned short* arow0 = pooled_bf + (size_t)(b * P + pbase + lm) * C;
  const unsigned short* arow1 = arow0 + (size_t)16 * C;
  const unsigned short* xTb   = xT_bf + (size_t)b * C * T;

  v8f acc0[8], acc1[8];            // [p-tile][c-tile]
#pragma unroll
  for (int j = 0; j < 8; ++j) { acc0[j] = {}; acc1[j] = {}; }

  for (int tb = 0; tb < T / 256; ++tb) {
    __syncthreads();   // previous phase2 done reading s_lds

    // ---- phase 1: S tiles (pp=0..1) x (tt = 2wv, 2wv+1), S = pooled@x^T ----
    v8f s00 = {}, s01 = {}, s10 = {}, s11 = {};
    const unsigned short* brow0 =
        x_bf + ((size_t)(b * T) + (size_t)(tb * 256 + wv * 32 + lm)) * C;
    const unsigned short* brow1 = brow0 + (size_t)16 * C;
#pragma unroll 2
    for (int kk = 0; kk < C / 32; ++kk) {
      ABOp a0, a1, b0, b1;
      const int c0 = kk * 32;
      a0.u2[0] = *(const v4u*)(arow0 + c0 + kbA);
      a0.u2[1] = *(const v4u*)(arow0 + c0 + 16 + kbA);
      a1.u2[0] = *(const v4u*)(arow1 + c0 + kbA);
      a1.u2[1] = *(const v4u*)(arow1 + c0 + 16 + kbA);
      b0.u2[0] = *(const v4u*)(brow0 + c0 + kbA);
      b0.u2[1] = *(const v4u*)(brow0 + c0 + 16 + kbA);
      b1.u2[0] = *(const v4u*)(brow1 + c0 + kbA);
      b1.u2[1] = *(const v4u*)(brow1 + c0 + 16 + kbA);
      s00 = __builtin_amdgcn_wmma_f32_16x16x32_bf16(false, a0.bf, false, b0.bf,
                                                    (short)0, s00, false, false);
      s01 = __builtin_amdgcn_wmma_f32_16x16x32_bf16(false, a0.bf, false, b1.bf,
                                                    (short)0, s01, false, false);
      s10 = __builtin_amdgcn_wmma_f32_16x16x32_bf16(false, a1.bf, false, b0.bf,
                                                    (short)0, s10, false, false);
      s11 = __builtin_amdgcn_wmma_f32_16x16x32_bf16(false, a1.bf, false, b1.bf,
                                                    (short)0, s11, false, false);
    }
    // D layout: VGPR r, lane -> (m = r + 8*hl, n = lm). Store bf16 S rows.
#pragma unroll
    for (int r = 0; r < 8; ++r) {
      const int prow = r + 8 * hl;
      const int tc0  = (wv * 2) * 16 + lm;
      s_lds[prow * SROW + tc0]             = f2bf(s00[r]);
      s_lds[prow * SROW + tc0 + 16]        = f2bf(s01[r]);
      s_lds[(prow + 16) * SROW + tc0]      = f2bf(s10[r]);
      s_lds[(prow + 16) * SROW + tc0 + 16] = f2bf(s11[r]);
    }
    __syncthreads();

    // ---- phase 2: acc[pp][j] += S(16x32 chunk) @ x(32 x 16c) via xT ----
    for (int kt = 0; kt < 8; ++kt) {
      ABOp a0, a1;
      a0.u2[0] = *(const v4u*)(s_lds + lm * SROW + kt * 32 + kbA);
      a0.u2[1] = *(const v4u*)(s_lds + lm * SROW + kt * 32 + 16 + kbA);
      a1.u2[0] = *(const v4u*)(s_lds + (lm + 16) * SROW + kt * 32 + kbA);
      a1.u2[1] = *(const v4u*)(s_lds + (lm + 16) * SROW + kt * 32 + 16 + kbA);
      const int tcol = tb * 256 + kt * 32;
#pragma unroll
      for (int j = 0; j < 8; ++j) {
        ABOp bb;
        const unsigned short* btr =
            xTb + (size_t)(wv * 128 + j * 16 + lm) * T + tcol;
        bb.u2[0] = *(const v4u*)(btr + kbA);
        bb.u2[1] = *(const v4u*)(btr + 16 + kbA);
        acc0[j] = __builtin_amdgcn_wmma_f32_16x16x32_bf16(
            false, a0.bf, false, bb.bf, (short)0, acc0[j], false, false);
        acc1[j] = __builtin_amdgcn_wmma_f32_16x16x32_bf16(
            false, a1.bf, false, bb.bf, (short)0, acc1[j], false, false);
      }
    }
  }

  // ---- epilogue: out = pooled + acc ----
#pragma unroll
  for (int j = 0; j < 8; ++j) {
    const int cg = wv * 128 + j * 16 + lm;
#pragma unroll
    for (int r = 0; r < 8; ++r) {
      const int pg0 = pbase + r + 8 * hl;
      const size_t o0 = (size_t)(b * P + pg0) * C + cg;
      out[o0] = pooled_f[o0] + acc0[j][r];
      const size_t o1 = o0 + (size_t)16 * C;
      out[o1] = pooled_f[o1] + acc1[j][r];
    }
  }
}

// ---------------------------------------------------------------------------
extern "C" void kernel_launch(void* const* d_in, const int* in_sizes, int n_in,
                              void* d_out, int out_size, void* d_ws, size_t ws_size,
                              hipStream_t stream) {
  const float* x = (const float*)d_in[0];
  float* out = (float*)d_out;

  // Workspace layout (88 MB total):
  //   [0,32M)   x_bf      bf16 row-major copy of x
  //   [32M,64M) xT_bf     bf16 transposed copy [b][c][t]
  //   [64M,80M) pooled_f  f32
  //   [80M,88M) pooled_bf bf16
  char* ws = (char*)d_ws;
  const size_t xBytes  = (size_t)Bb * T * C * 2;   // 32 MB
  const size_t pfBytes = (size_t)Bb * P * C * 4;   // 16 MB
  unsigned short* x_bf      = (unsigned short*)ws;
  unsigned short* xT_bf     = (unsigned short*)(ws + xBytes);
  float*          pooled_f  = (float*)(ws + 2 * xBytes);
  unsigned short* pooled_bf = (unsigned short*)(ws + 2 * xBytes + pfBytes);

  pool_kernel<<<(Bb * P * C) / 256, 256, 0, stream>>>(x, pooled_f, pooled_bf);
  cvt_transpose<<<dim3(T / 64, C / 64, Bb), 256, 0, stream>>>(x, x_bf, xT_bf);
  funnel_main<<<dim3(P / 32, Bb), 256, 0, stream>>>(pooled_f, pooled_bf, x_bf,
                                                    xT_bf, out);
}
